// FAAFusion_36275293782561
// MI455X (gfx1250) — compile-verified, matrix operand order checked by reading, not run on hardware
//
#include <hip/hip_runtime.h>
#include <hip/hip_bf16.h>
#include <math.h>

// ---------------------------------------------------------------------------
// FAA fusion pipeline for MI455X (gfx1250, wave32, WMMA)
//   pass 1: bilinear upsample x_high -> xhu                 (VALU, L2 resident)
//   pass 2: xl/xh channel projection 256->16 via WMMA bf16  (32x v_wmma / wave,
//           A-fragments preloaded once, 4 N-tiles per wave)
//   pass 3: per-patch 7x7 DFT magnitude argmax -> theta     (VALU, unrolled DFT)
//   pass 4: rotate(ph, t_low - t_high) + fold via atomics   (LDS patch staging)
//   pass 5: recon 16->256 via WMMA bf16 + fused epilogue    (4x v_wmma / wave)
// ---------------------------------------------------------------------------

typedef __attribute__((ext_vector_type(16))) __bf16 v16bf;
typedef __attribute__((ext_vector_type(8)))  float  v8f;

#define PI_F      3.14159265358979323846f
#define TWO_PI_F  6.28318530717958647692f
#define EPS_F     1e-8f

#define B_     2
#define C_IN   256
#define C_MID  16
#define H_     96
#define W_     96
#define HW     (H_*W_)       // 9216
#define HL_    48
#define WL_    48
#define M_     7
#define HO_    90
#define WO_    90
#define NPATCH (HO_*WO_)     // 8100
#define NT_    4             // N-tiles (of 16 pixels) per wave in GEMM kernels

// ------------------------------ pass 1: upsample ---------------------------
__global__ void k_upsample(const float* __restrict__ xh, float* __restrict__ out) {
  int tid = blockIdx.x * blockDim.x + threadIdx.x;
  const int total = B_ * C_IN * HW;
  if (tid >= total) return;
  int x  = tid % W_;
  int y  = (tid / W_) % H_;
  int bc = tid / HW;
  // align_corners=False bilinear, scale 2x: src = clamp((i+0.5)*0.5-0.5, 0, 47)
  float sy = fminf(fmaxf((y + 0.5f) * 0.5f - 0.5f, 0.0f), (float)(HL_ - 1));
  float sx = fminf(fmaxf((x + 0.5f) * 0.5f - 0.5f, 0.0f), (float)(WL_ - 1));
  int y0 = (int)floorf(sy); int y1 = min(y0 + 1, HL_ - 1); float wy = sy - (float)y0;
  int x0 = (int)floorf(sx); int x1 = min(x0 + 1, WL_ - 1); float wx = sx - (float)x0;
  const float* src = xh + (size_t)bc * HL_ * WL_;
  float r0 = src[y0 * WL_ + x0] * (1.0f - wx) + src[y0 * WL_ + x1] * wx;
  float r1 = src[y1 * WL_ + x0] * (1.0f - wx) + src[y1 * WL_ + x1] * wx;
  out[tid] = r0 * (1.0f - wy) + r1 * wy;
}

// --------------------- pass 2: projection GEMM via WMMA --------------------
// One wave computes a 16(C_MID) x 64(pixels) f32 strip: Y = W(16x256)*X(256x64).
// A (weight) fragments for all 8 K-tiles are loaded ONCE, then reused across
// 4 N-tiles -> 32 chained v_wmma_f32_16x16x32_bf16 per wave.
__global__ void k_proj_wmma(const float* __restrict__ x_low,
                            const float* __restrict__ xhu,
                            const float* __restrict__ w_low,
                            const float* __restrict__ w_high,
                            float* __restrict__ xl,
                            float* __restrict__ xh) {
  int wave = blockIdx.x * (blockDim.x >> 5) + (threadIdx.x >> 5);
  const int tgrps = HW / (16 * NT_);      // 144 tile-groups of 64 pixels
  const int total = 2 * B_ * tgrps;       // 576 waves
  if (wave >= total) return;
  int z   = wave / (B_ * tgrps);
  int r   = wave % (B_ * tgrps);
  int b   = r / tgrps;
  int pg0 = (r % tgrps) * (16 * NT_);

  const float* X  = (z == 0 ? x_low : xhu) + (size_t)b * C_IN * HW;
  const float* Wt = (z == 0 ? w_low : w_high);
  float*       Y  = (z == 0 ? xl : xh) + (size_t)b * C_MID * HW;

  int lane = threadIdx.x & 31;
  int mn   = lane & 15;   // A row (m) / B col (n)
  int kh   = lane >> 4;   // K-half selector per ISA lane layout

  // Preload A fragments (invariant across all N-tiles): 8 x v16bf = 64 VGPRs.
  v16bf afrag[8];
  #pragma unroll
  for (int kt = 0; kt < 8; ++kt) {
    int ks = kt * 32;
    #pragma unroll
    for (int e = 0; e < 16; ++e) {
      // lanes 0-15: K in {0..7, 16..23}; lanes 16-31: K in {8..15, 24..31}
      int k = ks + ((e < 8) ? (kh * 8 + e) : (16 + kh * 8 + (e - 8)));
      afrag[kt][e] = (__bf16)Wt[mn * C_IN + k];
    }
  }

  #pragma unroll
  for (int t = 0; t < NT_; ++t) {
    int pix0 = pg0 + t * 16;
    v8f acc = {0.f, 0.f, 0.f, 0.f, 0.f, 0.f, 0.f, 0.f};
    #pragma unroll
    for (int kt = 0; kt < 8; ++kt) {
      int ks = kt * 32;
      v16bf bm;
      #pragma unroll
      for (int e = 0; e < 16; ++e) {
        int k = ks + ((e < 8) ? (kh * 8 + e) : (16 + kh * 8 + (e - 8)));
        bm[e] = (__bf16)X[(size_t)k * HW + pix0 + mn];
      }
      if (kt < 7) __builtin_prefetch(&X[(size_t)(ks + 32) * HW + pix0 + mn], 0, 0);
      acc = __builtin_amdgcn_wmma_f32_16x16x32_bf16(false, afrag[kt], false, bm,
                                                    (short)0, acc, false, false);
    }
    #pragma unroll
    for (int j = 0; j < 8; ++j) {
      int m = j + 8 * kh;           // C/D layout: VGPR j holds row j + 8*(lane/16)
      Y[(size_t)m * HW + pix0 + mn] = acc[j];
    }
  }
}

// ---------------- pass 3: 7x7 DFT magnitude -> dominant direction ----------
__global__ void k_dir(const float* __restrict__ xl, const float* __restrict__ xh,
                      float* __restrict__ t_low, float* __restrict__ t_high) {
  int tid = blockIdx.x * blockDim.x + threadIdx.x;
  const int per = B_ * C_MID * NPATCH;   // 259200 patches per stream
  if (tid >= 2 * per) return;
  int z = tid / per;
  int g = tid % per;
  int b = g / (C_MID * NPATCH);
  int r = g % (C_MID * NPATCH);
  int c = r / NPATCH;
  int p = r % NPATCH;
  int py = p / WO_, px = p % WO_;

  const float* src = (z == 0 ? xl : xh) + (size_t)(b * C_MID + c) * HW;
  float* dst = (z == 0 ? t_low : t_high);

  float P[M_][M_];
  #pragma unroll
  for (int i = 0; i < M_; ++i)
    #pragma unroll
    for (int j = 0; j < M_; ++j)
      P[i][j] = src[(py + i) * W_ + (px + j)];

  // 7-entry twiddle table: e^{-2*pi*i*t/7}; exponents reduced mod 7 at compile time.
  float c7[M_], s7[M_];
  #pragma unroll
  for (int t = 0; t < M_; ++t) {
    float ang = -TWO_PI_F * (float)t / 7.0f;
    c7[t] = cosf(ang);
    s7[t] = sinf(ang);
  }

  // T = P * F^T  (real x complex)
  float Tre[M_][M_], Tim[M_][M_];
  #pragma unroll
  for (int j = 0; j < M_; ++j) {
    #pragma unroll
    for (int v = 0; v < M_; ++v) {
      float sre = 0.f, sim = 0.f;
      #pragma unroll
      for (int k = 0; k < M_; ++k) {
        int t = (v * k) % M_;
        sre = fmaf(P[j][k], c7[t], sre);
        sim = fmaf(P[j][k], s7[t], sim);
      }
      Tre[j][v] = sre; Tim[j][v] = sim;
    }
  }

  // S = F * T; argmax over (mag+eps)*rho, DC excluded; first-max tie break in
  // row-major (u,v) order to match the reference flatten order.
  float best = -1.0f, bth = 0.0f;
  #pragma unroll
  for (int u = 0; u < M_; ++u) {
    #pragma unroll
    for (int v = 0; v < M_; ++v) {
      if (u == 0 && v == 0) continue;
      float sre = 0.f, sim = 0.f;
      #pragma unroll
      for (int j = 0; j < M_; ++j) {
        int t = (u * j) % M_;
        sre += c7[t] * Tre[j][v] - s7[t] * Tim[j][v];
        sim += c7[t] * Tim[j][v] + s7[t] * Tre[j][v];
      }
      float fu  = (u <= 3) ? (float)u : (float)(u - M_);
      float fv  = (v <= 3) ? (float)v : (float)(v - M_);
      float rho = sqrtf(fu * fu + fv * fv);
      float mag = sqrtf(sre * sre + sim * sim) * (1.0f / 7.0f) + EPS_F; // ortho norm
      float score = mag * rho;
      if (score > best) {
        best = score;
        float th = atan2f(fu, fv);
        bth = fmodf(th + TWO_PI_F, TWO_PI_F);
      }
    }
  }
  dst[g] = fmodf(bth, PI_F);   // remainder(theta, pi), theta >= 0
}

// ------------------------------ pass 4a: zero fold -------------------------
__global__ void k_zero(float* __restrict__ buf, int n) {
  int tid = blockIdx.x * blockDim.x + threadIdx.x;
  if (tid < n) buf[tid] = 0.0f;
}

// ----------------- pass 4b: rotate patch + overlap-add fold ----------------
// Each thread owns one patch; its 7x7 source tile is staged in a private LDS
// slice so the bilinear gather's dynamic indexing hits LDS, not scratch.
__global__ void k_rotate_fold(const float* __restrict__ xh,
                              const float* __restrict__ t_low,
                              const float* __restrict__ t_high,
                              float* __restrict__ folded) {
  __shared__ float sp[128 * 50];   // 50-float slice per thread (pad past 49)
  int g = blockIdx.x * blockDim.x + threadIdx.x;
  const int total = B_ * C_MID * NPATCH;
  if (g >= total) return;
  int base = (int)threadIdx.x * 50;
  int b = g / (C_MID * NPATCH);
  int r = g % (C_MID * NPATCH);
  int c = r / NPATCH;
  int p = r % NPATCH;
  int py = p / WO_, px = p % WO_;

  const float* src = xh + (size_t)(b * C_MID + c) * HW;
  #pragma unroll
  for (int i = 0; i < M_; ++i)
    #pragma unroll
    for (int j = 0; j < M_; ++j)
      sp[base + i * M_ + j] = src[(py + i) * W_ + (px + j)];

  float theta = t_low[g] - t_high[g];
  float cs = cosf(theta), sn = sinf(theta);
  const float ctr = (float)(M_ - 1) * 0.5f;   // 3.0
  float tx = ctr - cs * ctr + sn * ctr;
  float ty = ctr - sn * ctr - cs * ctr;

  float* fdst = folded + (size_t)(b * C_MID + c) * HW;
  #pragma unroll
  for (int yy = 0; yy < M_; ++yy) {
    float yn = -1.0f + (2.0f * yy + 1.0f) / 7.0f;
    #pragma unroll
    for (int xx = 0; xx < M_; ++xx) {
      float xn = -1.0f + (2.0f * xx + 1.0f) / 7.0f;
      float gx = cs * xn - sn * yn + tx;
      float gy = sn * xn + cs * yn + ty;
      float ix = ((gx + 1.0f) * 7.0f - 1.0f) * 0.5f;
      float iy = ((gy + 1.0f) * 7.0f - 1.0f) * 0.5f;
      float x0 = floorf(ix), y0 = floorf(iy);
      float acc = 0.0f;
      #pragma unroll
      for (int dy = 0; dy < 2; ++dy) {
        #pragma unroll
        for (int dx = 0; dx < 2; ++dx) {
          float xc = x0 + (float)dx, yc = y0 + (float)dy;
          float wgt = (1.0f - fabsf(ix - xc)) * (1.0f - fabsf(iy - yc));
          bool valid = (xc >= 0.0f) && (xc <= 6.0f) && (yc >= 0.0f) && (yc <= 6.0f);
          int xi = (int)fminf(fmaxf(xc, 0.0f), 6.0f);
          int yi = (int)fminf(fmaxf(yc, 0.0f), 6.0f);
          float v = sp[base + yi * M_ + xi];
          acc += valid ? v * wgt : 0.0f;
        }
      }
      atomicAdd(&fdst[(py + yy) * W_ + (px + xx)], acc);
    }
  }
}

// --------------- pass 5: recon GEMM via WMMA + fused epilogue --------------
// rec(256xN) = Wr(256x16) * aligned(16xN); K padded 16->32 with zeros.
// A fragment loaded once per wave, reused across 4 N-tiles; epilogue
// out = x_low + layer_scale*rec + x_high_up fused into the D-tile store.
__global__ void k_recon_wmma(const float* __restrict__ folded,
                             const float* __restrict__ w_recon,
                             const float* __restrict__ x_low,
                             const float* __restrict__ xhu,
                             const float* __restrict__ ls_ptr,
                             float* __restrict__ out) {
  int wave = blockIdx.x * (blockDim.x >> 5) + (threadIdx.x >> 5);
  const int tgrps  = HW / (16 * NT_);       // 144
  const int mtiles = C_IN / 16;             // 16
  const int total  = B_ * mtiles * tgrps;   // 4608 waves
  if (wave >= total) return;
  int b   = wave / (mtiles * tgrps);
  int r   = wave % (mtiles * tgrps);
  int mt  = r / tgrps;
  int pg0 = (r % tgrps) * (16 * NT_);

  int lane = threadIdx.x & 31;
  int mn   = lane & 15;
  int kh   = lane >> 4;

  // A fragment: rows mt*16..mt*16+15 of w_recon, K 0..15 live, 16..31 zero.
  v16bf a;
  #pragma unroll
  for (int e = 0; e < 16; ++e) {
    int k = (e < 8) ? (kh * 8 + e) : (16 + kh * 8 + (e - 8));
    a[e] = (k < C_MID) ? (__bf16)w_recon[(mt * 16 + mn) * C_MID + k] : (__bf16)0.0f;
  }
  float ls = ls_ptr[0];
  const float* fsrc = folded + (size_t)b * C_MID * HW;

  #pragma unroll
  for (int t = 0; t < NT_; ++t) {
    int pix = pg0 + t * 16 + mn;
    int py = pix / W_, px = pix % W_;
    // overlap count: c1(a) = min(a+1, 7, 96-a)  (n-m+1 = 90 never binds first)
    float cy = fminf(fminf((float)(py + 1), 7.0f), (float)(H_ - py));
    float cx = fminf(fminf((float)(px + 1), 7.0f), (float)(W_ - px));
    float inv = 1.0f / (cy * cx + EPS_F);

    v16bf bm;
    #pragma unroll
    for (int e = 0; e < 16; ++e) {
      int k = (e < 8) ? (kh * 8 + e) : (16 + kh * 8 + (e - 8));
      bm[e] = (k < C_MID) ? (__bf16)(fsrc[(size_t)k * HW + pix] * inv) : (__bf16)0.0f;
    }
    v8f acc = {0.f, 0.f, 0.f, 0.f, 0.f, 0.f, 0.f, 0.f};
    acc = __builtin_amdgcn_wmma_f32_16x16x32_bf16(false, a, false, bm,
                                                  (short)0, acc, false, false);
    #pragma unroll
    for (int j = 0; j < 8; ++j) {
      int m = mt * 16 + j + 8 * kh;
      size_t oi = ((size_t)b * C_IN + m) * HW + pix;
      out[oi] = x_low[oi] + ls * acc[j] + xhu[oi];
    }
  }
}

// ---------------------------------------------------------------------------
extern "C" void kernel_launch(void* const* d_in, const int* in_sizes, int n_in,
                              void* d_out, int out_size, void* d_ws, size_t ws_size,
                              hipStream_t stream) {
  const float* x_high  = (const float*)d_in[0];
  const float* x_low   = (const float*)d_in[1];
  const float* w_low   = (const float*)d_in[2];
  const float* w_high  = (const float*)d_in[3];
  const float* w_recon = (const float*)d_in[4];
  const float* lscale  = (const float*)d_in[5];
  float* out = (float*)d_out;

  // workspace layout (floats); total ~24.5 MB, fully L2-resident on MI455X
  float* xhu    = (float*)d_ws;                             // 2*256*96*96
  float* xl     = xhu    + (size_t)B_ * C_IN  * HW;         // 2*16*96*96
  float* xh     = xl     + (size_t)B_ * C_MID * HW;         // 2*16*96*96
  float* tl     = xh     + (size_t)B_ * C_MID * HW;         // 2*16*8100
  float* th     = tl     + (size_t)B_ * C_MID * NPATCH;     // 2*16*8100
  float* folded = th     + (size_t)B_ * C_MID * NPATCH;     // 2*16*96*96

  {
    int total = B_ * C_IN * HW;
    k_upsample<<<(total + 255) / 256, 256, 0, stream>>>(x_high, xhu);
  }
  {
    // 576 waves (4 N-tiles each), 4 waves per 128-thread block
    k_proj_wmma<<<144, 128, 0, stream>>>(x_low, xhu, w_low, w_high, xl, xh);
  }
  {
    int total = 2 * B_ * C_MID * NPATCH;
    k_dir<<<(total + 127) / 128, 128, 0, stream>>>(xl, xh, tl, th);
  }
  {
    int total = B_ * C_MID * HW;
    k_zero<<<(total + 255) / 256, 256, 0, stream>>>(folded, total);
  }
  {
    int total = B_ * C_MID * NPATCH;
    k_rotate_fold<<<(total + 127) / 128, 128, 0, stream>>>(xh, tl, th, folded);
  }
  {
    // 4608 waves (4 N-tiles each), 4 per block
    k_recon_wmma<<<1152, 128, 0, stream>>>(folded, w_recon, x_low, xhu, lscale, out);
  }
}
